// DynamicSelfAttention_19825569038465
// MI455X (gfx1250) — matrix operand order repaired and checked by reading, not compiled, not run on hardware
//
#include <hip/hip_runtime.h>

// ---------------------------------------------------------------------------
// Problem constants (from reference)
// B=4, S=8, T=2048, QD=256, H=512; SCALE = (H/S)^-0.5 = 0.125
// ---------------------------------------------------------------------------
#define CB 4
#define CS 8
#define CT 2048
#define CQD 256
#define CH 512
#define CSCALE 0.125f

typedef __attribute__((ext_vector_type(16))) __bf16        v16bf;
typedef __attribute__((ext_vector_type(8)))  float         v8f;
typedef __attribute__((ext_vector_type(4)))  float         v4f;
typedef __attribute__((ext_vector_type(4)))  unsigned int  v4u;

union BF16x16 {
    unsigned int u[8];
    v4u          q[2];
    v16bf        v;
};

__device__ __forceinline__ unsigned short f2bf(float x) {
    unsigned int u = __float_as_uint(x);
    u += 0x7FFFu + ((u >> 16) & 1u);   // round-to-nearest-even
    return (unsigned short)(u >> 16);
}
__device__ __forceinline__ unsigned int pack2bf(float a, float b) {
    return (unsigned int)f2bf(a) | ((unsigned int)f2bf(b) << 16);
}
__device__ __forceinline__ v4u pack8bf(v4f lo, v4f hi) {
    v4u r = {pack2bf(lo.x, lo.y), pack2bf(lo.z, lo.w),
             pack2bf(hi.x, hi.y), pack2bf(hi.z, hi.w)};
    return r;
}

__device__ __forceinline__ v8f wmma_bf16(const v16bf& a, const v16bf& b, const v8f& c) {
    return __builtin_amdgcn_wmma_f32_16x16x32_bf16(false, a, false, b, (short)0, c,
                                                   false, false);
}

__device__ __forceinline__ float act_apply(float x, int ACT) {
    if (ACT == 1) return 0.5f * x * (1.0f + erff(x * 0.70710678118654752f)); // GELU
    if (ACT == 2) return x / (1.0f + __expf(-x));                            // SiLU
    return x;
}

// ---------------------------------------------------------------------------
// Weight prep: f32 [K][N] -> bf16 transposed [N][K] (tiled via LDS)
// ---------------------------------------------------------------------------
__global__ __launch_bounds__(256) void cvt_transpose_kernel(
    const float* __restrict__ in, unsigned short* __restrict__ out, int K, int N) {
    __shared__ unsigned short t[32][33];
    const int k0 = blockIdx.y * 32, n0 = blockIdx.x * 32;
    const int lx = threadIdx.x & 31, ly = threadIdx.x >> 5;   // 32 x 8
#pragma unroll
    for (int r = 0; r < 32; r += 8)
        t[ly + r][lx] = f2bf(in[(size_t)(k0 + ly + r) * N + n0 + lx]);
    __syncthreads();
#pragma unroll
    for (int r = 0; r < 32; r += 8)
        out[(size_t)(n0 + ly + r) * K + k0 + lx] = t[lx][ly + r];
}

// ---------------------------------------------------------------------------
// Generic bf16 WMMA GEMM:  Out[M,N] = act(A[M,K] @ W[K,N] + bias)
//   BM=128, BN=128, BK=32; 256 threads = 8 waves (2M x 4N), each wave 4x2
//   16x16 C-tiles. Double-buffered LDS ping-pong; chunk c+1 global loads in
//   flight during compute of chunk c; one barrier per chunk.
//   LDS rows padded to 40 elements (80 B, 16B-aligned, bank-stride 20):
//   every fragment = 2x ds_load_b128; all staging = b128 global + b128 DS.
//   FOLDED remaps A[r=(b,t), c=(s,h)] = src[((b*S+s)*T+t)*H + h].
// ---------------------------------------------------------------------------
template <int ACT, bool ABF16, bool OBF16, bool FOLDED>
__global__ __launch_bounds__(256) void gemm_bf16_kernel(
    const void* __restrict__ Araw, const unsigned short* __restrict__ Wt,
    const float* __restrict__ bias, void* __restrict__ Out,
    int M, int N, int K) {
    __shared__ __align__(16) unsigned short sA[2][128 * 40];
    __shared__ __align__(16) unsigned short sB[2][128 * 40];

    const int tid  = threadIdx.x;
    const int wave = tid >> 5;
    const int lane = tid & 31;
    const int wm   = wave >> 2;      // 0..1
    const int wn   = wave & 3;       // 0..3
    const int half = lane >> 4;
    const int l16  = lane & 15;
    const int row0 = blockIdx.y * 128;
    const int col0 = blockIdx.x * 128;
    const int NC   = K >> 5;

    v8f acc[4][2];
#pragma unroll
    for (int tm = 0; tm < 4; ++tm)
#pragma unroll
        for (int tn = 0; tn < 2; ++tn)
#pragma unroll
            for (int i = 0; i < 8; ++i) acc[tm][tn][i] = 0.0f;

    v4u ra[2], rb[2];
    // ---- one 32-k chunk: global -> regs, 16 B per op ----
    auto loadGlobal = [&](int c) {
        const int kb = c * 32;
#pragma unroll
        for (int i = 0; i < 2; ++i) {              // A: 512 v4u units
            int w  = i * 256 + tid;
            int r  = w >> 2;
            int kk = (w & 3) * 8;
            if (FOLDED) {
                int gr = row0 + r;
                int bb = gr >> 11;
                int tt = gr & (CT - 1);
                int cc = kb + kk;
                int ss = cc >> 9;
                int hh = cc & (CH - 1);
                size_t idx = (((size_t)bb * CS + ss) * CT + tt) * CH + hh;
                ra[i] = *(const v4u*)&((const unsigned short*)Araw)[idx];
            } else if (ABF16) {
                ra[i] = *(const v4u*)
                            &((const unsigned short*)Araw)[(size_t)(row0 + r) * K + kb + kk];
            } else {
                const float* src = (const float*)Araw + (size_t)(row0 + r) * K + kb + kk;
                ra[i] = pack8bf(*(const v4f*)src, *(const v4f*)(src + 4));
            }
        }
#pragma unroll
        for (int i = 0; i < 2; ++i) {              // B: 512 v4u units
            int w  = i * 256 + tid;
            int n  = w >> 2;
            int kk = (w & 3) * 8;
            rb[i] = *(const v4u*)&Wt[(size_t)(col0 + n) * K + kb + kk];
        }
    };
    auto storeLDS = [&](int buf) {
#pragma unroll
        for (int i = 0; i < 2; ++i) {
            int w  = i * 256 + tid;
            int r  = w >> 2;
            int kk = (w & 3) * 8;
            *(v4u*)&sA[buf][r * 40 + kk] = ra[i];
        }
#pragma unroll
        for (int i = 0; i < 2; ++i) {
            int w  = i * 256 + tid;
            int n  = w >> 2;
            int kk = (w & 3) * 8;
            *(v4u*)&sB[buf][n * 40 + kk] = rb[i];
        }
    };
    auto loadAfrag = [&](const unsigned short* pA, int tm, BF16x16& af) {
        int m = wm * 64 + tm * 16 + l16;
        af.q[0] = *(const v4u*)&pA[m * 40 + 8 * half];        // k = {0..7}+8h
        af.q[1] = *(const v4u*)&pA[m * 40 + 16 + 8 * half];   // k = {16..23}+8h
    };

    loadGlobal(0);
    storeLDS(0);
    __syncthreads();

    for (int c = 0; c < NC; ++c) {
        const int buf = c & 1;
        const unsigned short* pA = sA[buf];
        const unsigned short* pB = sB[buf];
        const bool more = (c + 1) < NC;
        if (more) loadGlobal(c + 1);               // overlaps with compute below

        BF16x16 bf0, bf1;
#pragma unroll
        for (int tn = 0; tn < 2; ++tn) {
            int n = wn * 32 + tn * 16 + l16;
            BF16x16& bf = tn ? bf1 : bf0;
            bf.q[0] = *(const v4u*)&pB[n * 40 + 8 * half];        // k = {0..7}+8h, {16..23}+8h
            bf.q[1] = *(const v4u*)&pB[n * 40 + 16 + 8 * half];
        }
        BF16x16 a0, a1;
        loadAfrag(pA, 0, a0);
#pragma unroll
        for (int tm = 0; tm < 4; tm += 2) {
            loadAfrag(pA, tm + 1, a1);
            acc[tm][0] = wmma_bf16(a0.v, bf0.v, acc[tm][0]);
            acc[tm][1] = wmma_bf16(a0.v, bf1.v, acc[tm][1]);
            if (tm + 2 < 4) loadAfrag(pA, tm + 2, a0);
            acc[tm + 1][0] = wmma_bf16(a1.v, bf0.v, acc[tm + 1][0]);
            acc[tm + 1][1] = wmma_bf16(a1.v, bf1.v, acc[tm + 1][1]);
        }
        if (more) storeLDS(buf ^ 1);               // other buffer: safe pre-barrier
        __syncthreads();
    }

    // ---- epilogue: bias + activation + store ----
#pragma unroll
    for (int tm = 0; tm < 4; ++tm) {
#pragma unroll
        for (int tn = 0; tn < 2; ++tn) {
#pragma unroll
            for (int vv = 0; vv < 8; ++vv) {
                int gr = row0 + wm * 64 + tm * 16 + vv + 8 * half;
                int gc = col0 + wn * 32 + tn * 16 + l16;
                float x = act_apply(acc[tm][tn][vv] + bias[gc], ACT);
                if (OBF16)
                    ((unsigned short*)Out)[(size_t)gr * N + gc] = f2bf(x);
                else
                    ((float*)Out)[(size_t)gr * N + gc] = x;
            }
        }
    }
}

// ---------------------------------------------------------------------------
// Flash attention over token axis T per (b, s):
//   out = softmax(qh kh^T * SCALE) vh + qh * alpha[s]   -> bf16
// 64 query rows / WG, 512 threads = 16 waves (mi 0..3 x ji 0..3).
// Q fragments in registers; online softmax; ~98 KB static LDS.
// K/Q rows padded to 520 (1040 B), V/P rows to 72 (144 B): all fragments are
// 2x ds_load_b128; staging is b128. global_prefetch of next K/V tiles.
// ---------------------------------------------------------------------------
__global__ __launch_bounds__(512) void flash_attn_kernel(
    const float* __restrict__ qh, const float* __restrict__ kh,
    const float* __restrict__ vh, const float* __restrict__ alpha,
    unsigned short* __restrict__ outbf) {
    __shared__ __align__(16) unsigned short sKV[36864];  // K:[64][520] / V:[512][72]
    __shared__ __align__(16) float          sS[64 * 66];
    __shared__ __align__(16) unsigned short sP[64 * 72];
    __shared__ float mS[64], lS[64], cS[64];

    const int tid  = threadIdx.x;
    const int wave = tid >> 5;
    const int lane = tid & 31;
    const int mi   = wave >> 2;
    const int ji   = wave & 3;
    const int half = lane >> 4;
    const int l16  = lane & 15;

    const int tok0 = blockIdx.x * 64;
    const int s    = blockIdx.y;
    const int b    = blockIdx.z;
    const size_t base = (((size_t)b * CS + s) * CT) * CH;

    // ---- stage Q (64 x 512) -> LDS (b128 ops) -> register fragments ----
#pragma unroll
    for (int i = 0; i < 4; ++i) {
        int w   = i * 512 + tid;          // 2048 v4u units
        int row = w >> 5;
        int dd  = (w & 31) * 8;
        const float* src = qh + base + (size_t)(tok0 + row) * CH + dd;
        *(v4u*)&sKV[row * 520 + dd] = pack8bf(*(const v4f*)src, *(const v4f*)(src + 4));
    }
    if (tid < 64) { mS[tid] = -1e30f; lS[tid] = 0.0f; }
    __syncthreads();

    BF16x16 qf[16];
#pragma unroll
    for (int kc = 0; kc < 16; ++kc) {
        int m = mi * 16 + l16;
        qf[kc].q[0] = *(const v4u*)&sKV[m * 520 + kc * 32 + 8 * half];
        qf[kc].q[1] = *(const v4u*)&sKV[m * 520 + kc * 32 + 16 + 8 * half];
    }
    __syncthreads();

    v8f acc[8];
#pragma unroll
    for (int t = 0; t < 8; ++t)
#pragma unroll
        for (int i = 0; i < 8; ++i) acc[t][i] = 0.0f;

    for (int it = 0; it < CT / 64; ++it) {
        const int kt0 = it * 64;

        // ---- load K tile [64 tok][512 d] -> sKV (b128 staging) ----
#pragma unroll
        for (int i = 0; i < 4; ++i) {
            int w   = i * 512 + tid;
            int row = w >> 5;
            int dd  = (w & 31) * 8;
            const float* src = kh + base + (size_t)(kt0 + row) * CH + dd;
            *(v4u*)&sKV[row * 520 + dd] = pack8bf(*(const v4f*)src, *(const v4f*)(src + 4));
        }
        __syncthreads();

        // prefetch next iteration's K/V tiles (256B granularity, full tiles)
        if (it + 1 < CT / 64) {
            const int prow = tid >> 3, pcol = (tid & 7) * 64;
            const size_t poff = base + (size_t)(kt0 + 64 + prow) * CH + pcol;
            __builtin_prefetch(&kh[poff], 0, 1);
            __builtin_prefetch(&vh[poff], 0, 1);
        }

        // ---- scores: S(mi,ji) = Q(mi) K(ji)^T * SCALE (pipelined frags) ----
        v8f sc;
#pragma unroll
        for (int i = 0; i < 8; ++i) sc[i] = 0.0f;
        {
            const int n = ji * 16 + l16;
            auto loadK = [&](int kc, BF16x16& bfr) {
                bfr.q[0] = *(const v4u*)&sKV[n * 520 + kc * 32 + 8 * half];
                bfr.q[1] = *(const v4u*)&sKV[n * 520 + kc * 32 + 16 + 8 * half];
            };
            BF16x16 b0, b1;
            loadK(0, b0);
#pragma unroll
            for (int kc = 0; kc < 16; kc += 2) {
                loadK(kc + 1, b1);
                sc = wmma_bf16(qf[kc].v, b0.v, sc);
                if (kc + 2 < 16) loadK(kc + 2, b0);
                sc = wmma_bf16(qf[kc + 1].v, b1.v, sc);
            }
        }
#pragma unroll
        for (int vv = 0; vv < 8; ++vv)
            sS[(mi * 16 + vv + 8 * half) * 66 + ji * 16 + l16] = sc[vv] * CSCALE;
        __syncthreads();

        // ---- online-softmax stats: 8 lanes per row (all in one wave32) ----
        {
            const int r = tid >> 3, sub = tid & 7;
            float mo = mS[r];
            float mx = mo;
            for (int j = sub; j < 64; j += 8) mx = fmaxf(mx, sS[r * 66 + j]);
#pragma unroll
            for (int off = 4; off > 0; off >>= 1) mx = fmaxf(mx, __shfl_xor(mx, off, 8));
            float sum = 0.0f;
            for (int j = sub; j < 64; j += 8) {
                float pj = __expf(sS[r * 66 + j] - mx);
                sum += pj;
                sP[r * 72 + j] = f2bf(pj);
            }
#pragma unroll
            for (int off = 4; off > 0; off >>= 1) sum += __shfl_xor(sum, off, 8);
            if (sub == 0) {
                float scl = __expf(mo - mx);
                mS[r] = mx;
                lS[r] = lS[r] * scl + sum;
                cS[r] = scl;
            }
        }
        // ---- load V tile transposed: sKV[d][tok], b128 = 8 tokens ----
#pragma unroll
        for (int j = 0; j < 8; ++j) {
            float t0 = vh[base + (size_t)(kt0 + 8 * j + 0) * CH + tid];
            float t1 = vh[base + (size_t)(kt0 + 8 * j + 1) * CH + tid];
            float t2 = vh[base + (size_t)(kt0 + 8 * j + 2) * CH + tid];
            float t3 = vh[base + (size_t)(kt0 + 8 * j + 3) * CH + tid];
            float t4 = vh[base + (size_t)(kt0 + 8 * j + 4) * CH + tid];
            float t5 = vh[base + (size_t)(kt0 + 8 * j + 5) * CH + tid];
            float t6 = vh[base + (size_t)(kt0 + 8 * j + 6) * CH + tid];
            float t7 = vh[base + (size_t)(kt0 + 8 * j + 7) * CH + tid];
            v4u pk = {pack2bf(t0, t1), pack2bf(t2, t3), pack2bf(t4, t5), pack2bf(t6, t7)};
            *(v4u*)&sKV[tid * 72 + 8 * j] = pk;
        }
        __syncthreads();

        // ---- rescale accumulators ----
        float s8[8];
#pragma unroll
        for (int vv = 0; vv < 8; ++vv) s8[vv] = cS[mi * 16 + vv + 8 * half];
#pragma unroll
        for (int t = 0; t < 8; ++t)
#pragma unroll
            for (int vv = 0; vv < 8; ++vv) acc[t][vv] *= s8[vv];

        // ---- O += P V (2 token chunks x 8 feature tiles, pipelined) ----
#pragma unroll
        for (int kc = 0; kc < 2; ++kc) {
            BF16x16 af;
            int m = mi * 16 + l16;
            af.q[0] = *(const v4u*)&sP[m * 72 + kc * 32 + 8 * half];
            af.q[1] = *(const v4u*)&sP[m * 72 + kc * 32 + 16 + 8 * half];
            auto loadV = [&](int t, BF16x16& bfr) {
                int n = ji * 128 + t * 16 + l16;
                bfr.q[0] = *(const v4u*)&sKV[n * 72 + kc * 32 + 8 * half];
                bfr.q[1] = *(const v4u*)&sKV[n * 72 + kc * 32 + 16 + 8 * half];
            };
            BF16x16 b0, b1;
            loadV(0, b0);
#pragma unroll
            for (int t = 0; t < 8; t += 2) {
                loadV(t + 1, b1);
                acc[t] = wmma_bf16(af.v, b0.v, acc[t]);
                if (t + 2 < 8) loadV(t + 2, b0);
                acc[t + 1] = wmma_bf16(af.v, b1.v, acc[t + 1]);
            }
        }
        __syncthreads();
    }

    // ---- epilogue: O/l + qh*alpha[s] -> bf16 ----
    const float al = alpha[s];
    float linv[8];
#pragma unroll
    for (int vv = 0; vv < 8; ++vv) linv[vv] = 1.0f / lS[mi * 16 + vv + 8 * half];
#pragma unroll
    for (int t = 0; t < 8; ++t) {
#pragma unroll
        for (int vv = 0; vv < 8; ++vv) {
            int lr = mi * 16 + vv + 8 * half;
            size_t idx = base + (size_t)(tok0 + lr) * CH + ji * 128 + t * 16 + l16;
            outbf[idx] = f2bf(acc[t][vv] * linv[vv] + qh[idx] * al);
        }
    }
}

// ---------------------------------------------------------------------------
// LayerNorm over last dim (512): one wave per row; f32 in -> bf16 out
// ---------------------------------------------------------------------------
__global__ __launch_bounds__(256) void layernorm_kernel(
    const float* __restrict__ x, const float* __restrict__ g,
    const float* __restrict__ bta, unsigned short* __restrict__ out) {
    const int row  = blockIdx.x * 8 + (threadIdx.x >> 5);
    const int lane = threadIdx.x & 31;
    const float* xr = x + (size_t)row * CH;
    float vals[16];
    float s = 0.0f, ss = 0.0f;
#pragma unroll
    for (int i = 0; i < 16; ++i) {
        float v = xr[lane + i * 32];
        vals[i] = v;
        s += v;
        ss += v * v;
    }
#pragma unroll
    for (int off = 16; off > 0; off >>= 1) {
        s  += __shfl_xor(s, off, 32);
        ss += __shfl_xor(ss, off, 32);
    }
    float mu   = s * (1.0f / CH);
    float var  = ss * (1.0f / CH) - mu * mu;
    float rinv = rsqrtf(var + 1e-5f);
    unsigned short* orow = out + (size_t)row * CH;
#pragma unroll
    for (int i = 0; i < 16; ++i) {
        int c = lane + i * 32;
        orow[c] = f2bf((vals[i] - mu) * rinv * g[c] + bta[c]);
    }
}

// ---------------------------------------------------------------------------
// Host-side orchestration (graph-capture safe: only stream launches)
// ---------------------------------------------------------------------------
extern "C" void kernel_launch(void* const* d_in, const int* in_sizes, int n_in,
                              void* d_out, int out_size, void* d_ws, size_t ws_size,
                              hipStream_t stream) {
    (void)in_sizes; (void)n_in; (void)out_size; (void)ws_size;

    const float* q    = (const float*)d_in[0];
    const float* k    = (const float*)d_in[1];
    const float* v    = (const float*)d_in[2];
    const float* Wq   = (const float*)d_in[3];  const float* bq   = (const float*)d_in[4];
    const float* Wk   = (const float*)d_in[5];  const float* bk   = (const float*)d_in[6];
    const float* Wv   = (const float*)d_in[7];  const float* bv   = (const float*)d_in[8];
    const float* alp  = (const float*)d_in[9];
    const float* mhoW = (const float*)d_in[10]; const float* mhob = (const float*)d_in[11];
    const float* lng  = (const float*)d_in[12]; const float* lnb  = (const float*)d_in[13];
    const float* ffW1 = (const float*)d_in[14]; const float* ffb1 = (const float*)d_in[15];
    const float* ffW2 = (const float*)d_in[16]; const float* ffb2 = (const float*)d_in[17];
    const float* soW1 = (const float*)d_in[18]; const float* sob1 = (const float*)d_in[19];
    const float* soW2 = (const float*)d_in[20]; const float* sob2 = (const float*)d_in[21];

    const size_t Mtok = (size_t)CB * CS * CT;   // 65536
    const int    Mseq = CB * CT;                // 8192

    // ---- workspace carve ----
    char* p = (char*)d_ws;
    auto carve = [&](size_t bytes) -> void* {
        void* r = (void*)p;
        p += (bytes + 255) & ~(size_t)255;
        return r;
    };
    unsigned short* wqb   = (unsigned short*)carve((size_t)CQD * CH * 2);
    unsigned short* wkb   = (unsigned short*)carve((size_t)CQD * CH * 2);
    unsigned short* wvb   = (unsigned short*)carve((size_t)CQD * CH * 2);
    unsigned short* wmhob = (unsigned short*)carve((size_t)CH * CH * 2);
    unsigned short* wff1b = (unsigned short*)carve((size_t)CH * 2 * CH * 2);
    unsigned short* wff2b = (unsigned short*)carve((size_t)2 * CH * CH * 2);
    unsigned short* wso1b = (unsigned short*)carve((size_t)CS * CH * CH * 2);
    unsigned short* wso2b = (unsigned short*)carve((size_t)CH * CH * 2);
    float* qhf = (float*)carve(Mtok * CH * 4);
    float* khf = (float*)carve(Mtok * CH * 4);
    float* vhf = (float*)carve(Mtok * CH * 4);
    unsigned short* attnb = (unsigned short*)carve(Mtok * CH * 2);
    // aliases (stream order makes these safe):
    float*          mhof = qhf;                   // after flash consumed qh
    unsigned short* lnbf = (unsigned short*)khf;  // after flash
    unsigned short* ff1b = (unsigned short*)vhf;  // after flash
    unsigned short* ff2b = attnb;                 // after mho GEMM
    unsigned short* so1b = (unsigned short*)qhf;  // after LN consumed mho

    // ---- weights -> bf16, transposed to [N][K] ----
    auto cvtT = [&](const float* src, unsigned short* dst, int K, int N) {
        cvt_transpose_kernel<<<dim3(N / 32, K / 32), 256, 0, stream>>>(src, dst, K, N);
    };
    cvtT(Wq,   wqb,   CQD,    CH);
    cvtT(Wk,   wkb,   CQD,    CH);
    cvtT(Wv,   wvb,   CQD,    CH);
    cvtT(mhoW, wmhob, CH,     CH);
    cvtT(ffW1, wff1b, CH,     2 * CH);
    cvtT(ffW2, wff2b, 2 * CH, CH);
    cvtT(soW1, wso1b, CS * CH, CH);
    cvtT(soW2, wso2b, CH,     CH);

    const dim3 blk(256);
    // ---- projections: [65536,256] @ [256,512] -> f32 ----
    {
        dim3 g(CH / 128, (int)(Mtok / 128));
        gemm_bf16_kernel<0, false, false, false><<<g, blk, 0, stream>>>(q, wqb, bq, qhf, (int)Mtok, CH, CQD);
        gemm_bf16_kernel<0, false, false, false><<<g, blk, 0, stream>>>(k, wkb, bk, khf, (int)Mtok, CH, CQD);
        gemm_bf16_kernel<0, false, false, false><<<g, blk, 0, stream>>>(v, wvb, bv, vhf, (int)Mtok, CH, CQD);
    }
    // ---- flash attention + epilogue -> bf16 ----
    flash_attn_kernel<<<dim3(CT / 64, CS, CB), 512, 0, stream>>>(qhf, khf, vhf, alp, attnb);
    // ---- multi_head_out linear: [65536,512]@[512,512] -> f32 ----
    gemm_bf16_kernel<0, true, false, false><<<dim3(CH / 128, (int)(Mtok / 128)), blk, 0, stream>>>(
        attnb, wmhob, mhob, mhof, (int)Mtok, CH, CH);
    // ---- layernorm -> bf16 ----
    layernorm_kernel<<<(int)(Mtok / 8), 256, 0, stream>>>(mhof, lng, lnb, lnbf);
    // ---- FF: gelu(x@W1+b1)@W2+b2 ----
    gemm_bf16_kernel<1, true, true, false><<<dim3(2 * CH / 128, (int)(Mtok / 128)), blk, 0, stream>>>(
        lnbf, wff1b, ffb1, ff1b, (int)Mtok, 2 * CH, CH);
    gemm_bf16_kernel<0, true, true, false><<<dim3(CH / 128, (int)(Mtok / 128)), blk, 0, stream>>>(
        ff1b, wff2b, ffb2, ff2b, (int)Mtok, CH, 2 * CH);
    // ---- seq_out: fold [B,S,T,H] -> [B,T,S*H]; silu(x@W1+b1)@W2+b2 ----
    gemm_bf16_kernel<2, true, true, true><<<dim3(CH / 128, Mseq / 128), blk, 0, stream>>>(
        ff2b, wso1b, sob1, so1b, Mseq, CH, CS * CH);
    gemm_bf16_kernel<0, true, false, false><<<dim3(CH / 128, Mseq / 128), blk, 0, stream>>>(
        so1b, wso2b, sob2, (float*)d_out, Mseq, CH, CH);
}
